// ContinuousFilterConvolution_20753281974333
// MI455X (gfx1250) — compile-verified, matrix operand order checked by reading, not run on hardware
//
#include <hip/hip_runtime.h>

// ContinuousFilterConvolution for MI455X (gfx1250, wave32, WMMA).
//
// Roofline: rbf_expansion read = 268 MB -> ~11.5us floor @ 23.3 TB/s.
// 51.5 GFLOP of chained GEMMs -> must go through v_wmma_f32_16x16x32_f16,
// scalar f32 VALU would be >10x slower than the memory floor.
//
// Transposed formulation: h^T = ssp(W1^T @ rbf^T + b1), filt^T = W2^T @ h^T + b2.
// GEMM1's f32 D-tile layout (VGPR r, lane L -> M=r+(L>=16?8:0), N=L&15) maps
// lane-locally onto GEMM2's f16 B-operand layout (lane L -> col N=L&15,
// elements 0..7 = K-chunk low half (off by 8 for lanes>=16), 8..15 = +16),
// so the inter-GEMM ssp + f32->f16 pack needs NO cross-lane movement.

typedef __attribute__((ext_vector_type(16))) _Float16 v16h;
typedef __attribute__((ext_vector_type(8)))  float    v8f;
typedef __attribute__((ext_vector_type(4)))  float    v4f;

#define BDIM 32
#define NDIM 512
#define KDIM 64
#define GDIM 64
#define FDIM 128

__device__ __forceinline__ float ssp(float x) {
    // shifted softplus: log(1+exp(x)) - log(2), overflow-safe
    float ax = __builtin_fabsf(x);
    return fmaxf(x, 0.0f) + __logf(1.0f + __expf(-ax)) - 0.69314718056f;
}

__global__ __launch_bounds__(256, 1)
void cfconv_wmma_kernel(const float* __restrict__ features,      // [B,N,F]
                        const float* __restrict__ rbf,           // [B,N,K,G]
                        const int*   __restrict__ nbr,           // [B,N,K]
                        const float* __restrict__ mask,          // [B,N,K]
                        const float* __restrict__ W1,            // [G,F]
                        const float* __restrict__ b1,            // [F]
                        const float* __restrict__ W2,            // [F,F]
                        const float* __restrict__ b2,            // [F]
                        float*       __restrict__ out) {         // [B,N,F]
    // A-operand layout staged in LDS (f16): slot = (tile, chunk, lane) -> v16h.
    __shared__ v16h ldsW1[8 * 2 * 32];   // W1^T: 8 f-tiles x 2 K-chunks(G=64)  : 16 KB
    __shared__ v16h ldsW2[8 * 4 * 32];   // W2^T: 8 o-tiles x 4 K-chunks(F=128) : 32 KB

    const int tid = threadIdx.x;

    // ---- pack W1^T into WMMA A layout: lane L holds row f=16*tile+(L&15),
    //      element e -> g = 32*chunk + (e<8 ? off+e : 16+off+e-8), off=(L&16)?8:0
    for (int slot = tid; slot < 8 * 2 * 32; slot += 256) {
        int lane  = slot & 31;
        int chunk = (slot >> 5) & 1;
        int tile  = slot >> 6;
        int off   = (lane & 16) ? 8 : 0;
        int f     = 16 * tile + (lane & 15);
        v16h v;
#pragma unroll
        for (int e = 0; e < 16; ++e) {
            int gl = (e < 8) ? (off + e) : (16 + off + (e - 8));
            v[e] = (_Float16)W1[(32 * chunk + gl) * FDIM + f];
        }
        ldsW1[slot] = v;
    }
    // ---- pack W2^T likewise (K dim = F = 128, 4 chunks)
    for (int slot = tid; slot < 8 * 4 * 32; slot += 256) {
        int lane  = slot & 31;
        int chunk = (slot >> 5) & 3;
        int tile  = slot >> 7;
        int off   = (lane & 16) ? 8 : 0;
        int o     = 16 * tile + (lane & 15);
        v16h v;
#pragma unroll
        for (int e = 0; e < 16; ++e) {
            int fl = (e < 8) ? (off + e) : (16 + off + (e - 8));
            v[e] = (_Float16)W2[(32 * chunk + fl) * FDIM + o];
        }
        ldsW2[slot] = v;
    }
    __syncthreads();

    const int lane = tid & 31;
    const int wave = tid >> 5;
    const int pair = blockIdx.x * 8 + wave;          // pair = b*N + n
    const int b    = pair >> 9;                      // N = 512
    const int off  = (lane & 16) ? 8 : 0;            // K/M half select
    const int row  = lane & 15;                      // neighbor-row within tile

    const float* rbf_bn  = rbf  + (size_t)pair * KDIM * GDIM;
    const int*   nbr_bn  = nbr  + (size_t)pair * KDIM;
    const float* mask_bn = mask + (size_t)pair * KDIM;
    const float* feat_b  = features + (size_t)b * NDIM * FDIM;

    float acc[8][8];
#pragma unroll
    for (int ot = 0; ot < 8; ++ot)
#pragma unroll
        for (int r = 0; r < 8; ++r) acc[ot][r] = 0.0f;

    for (int kt = 0; kt < 4; ++kt) {                 // 4 tiles of 16 neighbor rows
        const int k = kt * 16 + row;
        const float* rbf_k = rbf_bn + (size_t)k * GDIM;
        const int    j     = nbr_bn[k];
        const float  m     = mask_bn[k];
        const float* feat_j = feat_b + (size_t)j * FDIM + off;

        v16h hb[4];                                  // h^T as GEMM2 B-operand chunks

        // ---- GEMM1': h^T tile = ssp(W1^T @ rbf^T + b1) per 16-wide f-tile
#pragma unroll
        for (int ft = 0; ft < 8; ++ft) {
            v8f d = {};
#pragma unroll
            for (int c = 0; c < 2; ++c) {
                // B-operand: rbf^T chunk, lane = neighbor row, elements = g values
                const float* p = rbf_k + 32 * c + off;
                v4f lo0 = *(const v4f*)(p);
                v4f lo1 = *(const v4f*)(p + 4);
                v4f hi0 = *(const v4f*)(p + 16);
                v4f hi1 = *(const v4f*)(p + 20);
                v16h bm;
#pragma unroll
                for (int e = 0; e < 4; ++e) {
                    bm[e]      = (_Float16)lo0[e];
                    bm[e + 4]  = (_Float16)lo1[e];
                    bm[e + 8]  = (_Float16)hi0[e];
                    bm[e + 12] = (_Float16)hi1[e];
                }
                v16h am = ldsW1[(ft * 2 + c) * 32 + lane];
                d = __builtin_amdgcn_wmma_f32_16x16x32_f16(
                        false, am, false, bm, (short)0, d, false, false);
            }
            // bias + ssp, pack lane-locally into GEMM2 B-chunk (ft>>1),
            // element offset 0 for even ft, 8 for odd ft.
            const float* b1p  = b1 + 16 * ft + off;
            const int    eoff = (ft & 1) ? 8 : 0;
#pragma unroll
            for (int r = 0; r < 8; ++r)
                hb[ft >> 1][eoff + r] = (_Float16)ssp(d[r] + b1p[r]);
        }

        // ---- GEMM2': filt^T = W2^T @ h^T + b2, fused gather/mask/accumulate
#pragma unroll
        for (int ot = 0; ot < 8; ++ot) {
            v8f d = {};
#pragma unroll
            for (int c = 0; c < 4; ++c) {
                v16h am = ldsW2[(ot * 4 + c) * 32 + lane];
                d = __builtin_amdgcn_wmma_f32_16x16x32_f16(
                        false, am, false, hb[c], (short)0, d, false, false);
            }
            const float* b2p = b2 + 16 * ot + off;
            const float* nfp = feat_j + 16 * ot;     // gathered neighbor features
            v4f nf0 = *(const v4f*)(nfp);
            v4f nf1 = *(const v4f*)(nfp + 4);
#pragma unroll
            for (int r = 0; r < 8; ++r) {
                float nf = (r < 4) ? nf0[r] : nf1[r - 4];
                acc[ot][r] += (d[r] + b2p[r]) * nf * m;
            }
        }
    }

    // ---- sum over the 16 neighbor rows: lanes 0..15 and 16..31 reduce
    //      independently (xor masks 1,2,4,8 never cross the 16-lane boundary)
    float* out_bn = out + (size_t)pair * FDIM;
#pragma unroll
    for (int ot = 0; ot < 8; ++ot) {
#pragma unroll
        for (int r = 0; r < 8; ++r) {
            float v = acc[ot][r];
            v += __shfl_xor(v, 1, 32);
            v += __shfl_xor(v, 2, 32);
            v += __shfl_xor(v, 4, 32);
            v += __shfl_xor(v, 8, 32);
            acc[ot][r] = v;
        }
        if (row == 0) {                              // lanes 0 and 16 store
#pragma unroll
            for (int r = 0; r < 8; ++r)
                out_bn[16 * ot + off + r] = acc[ot][r];
        }
    }
}

extern "C" void kernel_launch(void* const* d_in, const int* in_sizes, int n_in,
                              void* d_out, int out_size, void* d_ws, size_t ws_size,
                              hipStream_t stream) {
    const float* features = (const float*)d_in[0];   // [32,512,128]
    const float* rbf      = (const float*)d_in[1];   // [32,512,64,64]
    const int*   nbr      = (const int*)  d_in[2];   // [32,512,64]
    const float* mask     = (const float*)d_in[3];   // [32,512,64]
    const float* W1       = (const float*)d_in[4];   // [64,128]
    const float* b1       = (const float*)d_in[5];   // [128]
    const float* W2       = (const float*)d_in[6];   // [128,128]
    const float* b2       = (const float*)d_in[7];   // [128]
    float*       out      = (float*)d_out;           // [32,512,128]

    const int pairs  = BDIM * NDIM;                  // 16384 (b,n) pairs
    const int blocks = pairs / 8;                    // 8 waves (pairs) per block
    cfconv_wmma_kernel<<<blocks, 256, 0, stream>>>(features, rbf, nbr, mask,
                                                   W1, b1, W2, b2, out);
}
